// GraphAttentionEmbedding_44616120271327
// MI455X (gfx1250) — matrix operand order repaired and checked by reading, not compiled, not run on hardware
//
#include <hip/hip_runtime.h>
#include <math.h>

typedef __attribute__((ext_vector_type(16))) __bf16 v16bf;
typedef __attribute__((ext_vector_type(8)))  __bf16 v8bf;
typedef __attribute__((ext_vector_type(8)))  float  v8f;

#define KNBR  10
#define DFEAT 256

#define F_BF16   1
#define F_RELU   2
#define F_ROWMUL 4

// ---------------------------------------------------------------------------
// small elementwise / gather kernels
// ---------------------------------------------------------------------------

__global__ void k_te0(const float* __restrict__ time_b, float* __restrict__ te0) {
  int t = threadIdx.x;
  te0[t] = cosf(time_b[t]);
}

// src: [Kd, N] f32 row-major  ->  dst: [N, Kd] bf16 row-major (transposed)
__global__ void k_convert_wT(const float* __restrict__ src, __bf16* __restrict__ dst,
                             int Kd, int N) {
  int total = Kd * N;
  for (int tid = blockIdx.x * blockDim.x + threadIdx.x; tid < total;
       tid += gridDim.x * blockDim.x) {
    int n = tid / Kd;
    int k = tid - n * Kd;
    dst[tid] = (__bf16)src[(size_t)k * N + n];
  }
}

// out[r,:] = memory[idx[r],:] + nodef[idx[r],:]
__global__ void k_gather_feats(const float* __restrict__ memory,
                               const float* __restrict__ nodef,
                               const int* __restrict__ idx,
                               float* __restrict__ out, int nRows) {
  int r = blockIdx.x;
  if (r >= nRows) return;
  int c = threadIdx.x;
  size_t base = (size_t)idx[r] * DFEAT;
  out[(size_t)r * DFEAT + c] = memory[base + c] + nodef[base + c];
}

// qin[r] = [ bf16(srcfeat[r,:256]) | bf16(te0) ]   (512 cols)
__global__ void k_build_qin(const float* __restrict__ srcfeat,
                            const float* __restrict__ te0,
                            __bf16* __restrict__ qin) {
  int r = blockIdx.x;
  int c = threadIdx.x;
  qin[(size_t)r * 512 + c]       = (__bf16)srcfeat[(size_t)r * DFEAT + c];
  qin[(size_t)r * 512 + 256 + c] = (__bf16)te0[c];
}

// kin[r] = [ nbr_emb | cos(dt*w+b) | edge_feat ]   (768 cols), bf16
__global__ void k_build_kin(const float* __restrict__ memory,
                            const float* __restrict__ nodef,
                            const float* __restrict__ edgef,
                            const float* __restrict__ time_w,
                            const float* __restrict__ time_b,
                            const float* __restrict__ direct_rows,
                            const int* __restrict__ nbr_idx,
                            const int* __restrict__ edge_idx,
                            const float* __restrict__ timestamps,
                            const float* __restrict__ edge_times,
                            int tsDiv, int mode, int rowBase,
                            __bf16* __restrict__ kin) {
  int r = blockIdx.x;
  int g = rowBase + r;            // global key row
  int c = threadIdx.x;
  int q = g / KNBR;               // query index
  float ts = timestamps[q / tsDiv];
  float dt = ts - edge_times[g];
  float v0;
  if (mode == 0) {
    size_t nb = (size_t)nbr_idx[g] * DFEAT;
    v0 = memory[nb + c] + nodef[nb + c];
  } else {
    v0 = direct_rows[(size_t)g * DFEAT + c];
  }
  size_t ro = (size_t)r * 768;
  kin[ro + c]       = (__bf16)v0;
  kin[ro + 256 + c] = (__bf16)cosf(dt * time_w[c] + time_b[c]);
  kin[ro + 512 + c] = (__bf16)edgef[(size_t)edge_idx[g] * DFEAT + c];
}

// Min[:,512:768] = bf16(srcfeat)
__global__ void k_fill_min_src(const float* __restrict__ srcfeat,
                               __bf16* __restrict__ minbuf) {
  int r = blockIdx.x; int c = threadIdx.x;
  minbuf[(size_t)r * 768 + 512 + c] = (__bf16)srcfeat[(size_t)r * DFEAT + c];
}

// per (query m, head h): masked softmax over K=10 keys, weighted V sum -> bf16 o
__global__ void k_attn_combine(const float* __restrict__ Qh,
                               const float* __restrict__ Kh,
                               const float* __restrict__ Vh,
                               const int* __restrict__ nbr_idx, int keyBase,
                               __bf16* __restrict__ obf, float* __restrict__ oMul,
                               int nQ) {
  int t = blockIdx.x * blockDim.x + threadIdx.x;
  if (t >= nQ * 8) return;
  int m = t >> 3, h = t & 7;
  const float* qp = Qh + (size_t)m * 512 + h * 64;
  float s[KNBR];
  bool allm = true;
  for (int k = 0; k < KNBR; ++k) {
    int nb = nbr_idx[keyBase + m * KNBR + k];
    if (nb == 0) {
      s[k] = -1e9f;
    } else {
      allm = false;
      const float* kp = Kh + (size_t)(m * KNBR + k) * 512 + h * 64;
      float acc = 0.f;
      #pragma unroll 8
      for (int d = 0; d < 64; ++d) acc += qp[d] * kp[d];
      s[k] = acc * 0.125f;   // / sqrt(64)
    }
  }
  float mx = s[0];
  for (int k = 1; k < KNBR; ++k) mx = fmaxf(mx, s[k]);
  float sum = 0.f;
  for (int k = 0; k < KNBR; ++k) { s[k] = __expf(s[k] - mx); sum += s[k]; }
  float inv = 1.f / sum;
  if (h == 0) oMul[m] = allm ? 0.f : 1.f;
  __bf16* op = obf + (size_t)m * 512 + h * 64;
  for (int d = 0; d < 64; ++d) {
    float acc = 0.f;
    for (int k = 0; k < KNBR; ++k)
      acc += s[k] * Vh[(size_t)(m * KNBR + k) * 512 + h * 64 + d];
    op[d] = (__bf16)(acc * inv);
  }
}

// ---------------------------------------------------------------------------
// WMMA bf16 GEMM:  out[M,N] = A[M,Kd] @ W[Kd,N] + bias  (W given transposed)
// workgroup = 8 waves -> 128x128 C block; wave = 32 rows x 64 cols (8 tiles)
// ---------------------------------------------------------------------------
__global__ void __launch_bounds__(256) k_gemm_bf16(
    const __bf16* __restrict__ A, int lda,
    const __bf16* __restrict__ WT, int Kd,      // WT: [N, Kd] row-major
    const float* __restrict__ bias,
    const float* __restrict__ rowMul,
    float* __restrict__ outF, __bf16* __restrict__ outH,
    int ldout, int flags) {
  int wave = threadIdx.x >> 5;
  int lane = threadIdx.x & 31;
  int ln = lane & 15, hi = lane >> 4;
  int R = blockIdx.y * 128 + (wave & 3) * 32;   // wave row base
  int C = blockIdx.x * 128 + (wave >> 2) * 64;  // wave col base

  v8f zero = {0.f, 0.f, 0.f, 0.f, 0.f, 0.f, 0.f, 0.f};
  v8f acc[2][4];
  #pragma unroll
  for (int rt = 0; rt < 2; ++rt)
    #pragma unroll
    for (int ct = 0; ct < 4; ++ct) acc[rt][ct] = zero;

  for (int k0 = 0; k0 < Kd; k0 += 32) {
    // A fragments: 16x32 bf16; lane(ln,hi): j<8 -> K=k0+hi*8+j, j>=8 -> +16
    v16bf a[2];
    #pragma unroll
    for (int rt = 0; rt < 2; ++rt) {
      const __bf16* ap = A + (size_t)(R + rt * 16 + ln) * lda + k0 + hi * 8;
      v8bf lo = *(const v8bf*)ap;
      v8bf hh = *(const v8bf*)(ap + 16);
      #pragma unroll
      for (int j = 0; j < 8; ++j) { a[rt][j] = lo[j]; a[rt][j + 8] = hh[j]; }
    }
    #pragma unroll
    for (int ct = 0; ct < 4; ++ct) {
      // B fragment: 32x16; lane holds column (C+ct*16+ln), j<8 -> K=k0+hi*8+j, j>=8 -> +16
      const __bf16* bp = WT + (size_t)(C + ct * 16 + ln) * Kd + k0 + hi * 8;
      v8bf b0 = *(const v8bf*)bp;
      v8bf b1 = *(const v8bf*)(bp + 16);
      v16bf b;
      #pragma unroll
      for (int j = 0; j < 8; ++j) { b[j] = b0[j]; b[j + 8] = b1[j]; }
      #pragma unroll
      for (int rt = 0; rt < 2; ++rt)
        acc[rt][ct] = __builtin_amdgcn_wmma_f32_16x16x32_bf16(
            false, a[rt], false, b, (short)0, acc[rt][ct], false, false);
    }
  }

  // epilogue: C/D layout — lane col = ln, row = hi*8 + r (per 16x16 tile)
  #pragma unroll
  for (int ct = 0; ct < 4; ++ct) {
    int col = C + ct * 16 + ln;
    float bv = bias[col];
    #pragma unroll
    for (int rt = 0; rt < 2; ++rt) {
      #pragma unroll
      for (int rr = 0; rr < 8; ++rr) {
        int row = R + rt * 16 + hi * 8 + rr;
        float v = acc[rt][ct][rr] + bv;
        if (flags & F_ROWMUL) v *= rowMul[row];
        if (flags & F_RELU)   v = fmaxf(v, 0.f);
        size_t o = (size_t)row * ldout + col;
        if (flags & F_BF16) outH[o] = (__bf16)v;
        else                outF[o] = v;
      }
    }
  }
}

// ---------------------------------------------------------------------------
extern "C" void kernel_launch(void* const* d_in, const int* in_sizes, int n_in,
                              void* d_out, int out_size, void* d_ws, size_t ws_size,
                              hipStream_t stream) {
  (void)in_sizes; (void)n_in; (void)out_size; (void)ws_size;
  const float* memory   = (const float*)d_in[0];
  const float* nodef    = (const float*)d_in[1];
  const float* edgef    = (const float*)d_in[2];
  const float* time_w   = (const float*)d_in[3];
  const float* time_b   = (const float*)d_in[4];
  const float* Wq = (const float*)d_in[5];  const float* bq = (const float*)d_in[6];
  const float* Wk = (const float*)d_in[7];  const float* bk = (const float*)d_in[8];
  const float* Wv = (const float*)d_in[9];  const float* bv = (const float*)d_in[10];
  const float* Wo = (const float*)d_in[11]; const float* bo = (const float*)d_in[12];
  const float* W1 = (const float*)d_in[13]; const float* b1 = (const float*)d_in[14];
  const float* W2 = (const float*)d_in[15]; const float* b2 = (const float*)d_in[16];
  const float* timestamps  = (const float*)d_in[17];
  const float* edge_times1 = (const float*)d_in[18];
  const float* edge_times2 = (const float*)d_in[19];
  const int* src_nodes  = (const int*)d_in[20];
  const int* neighbors1 = (const int*)d_in[21];
  const int* edge_idx1  = (const int*)d_in[22];
  const int* neighbors2 = (const int*)d_in[23];
  const int* edge_idx2  = (const int*)d_in[24];
  (void)src_nodes;

  const int B = 2048, K = KNBR, CH = 1024, CHK = CH * K;

  // ---- workspace carve-up (256B aligned) ----
  uint8_t* base = (uint8_t*)d_ws;
  size_t off = 0;
  auto carve = [&](size_t bytes) -> void* {
    void* p = base + off;
    off = (off + bytes + 255) & ~(size_t)255;
    return p;
  };
  __bf16 *WqT[2], *WkT[2], *WvT[2], *WoT[2], *W1T[2], *W2T[2];
  for (int l = 0; l < 2; ++l) {
    WqT[l] = (__bf16*)carve((size_t)512 * 512 * 2);
    WkT[l] = (__bf16*)carve((size_t)512 * 768 * 2);
    WvT[l] = (__bf16*)carve((size_t)512 * 768 * 2);
    WoT[l] = (__bf16*)carve((size_t)512 * 512 * 2);
    W1T[l] = (__bf16*)carve((size_t)256 * 768 * 2);
    W2T[l] = (__bf16*)carve((size_t)256 * 256 * 2);
  }
  float*  te0    = (float*)carve(256 * 4);
  float*  f_src0 = (float*)carve((size_t)B * 256 * 4);
  float*  f_n1_0 = (float*)carve((size_t)B * K * 256 * 4);
  float*  src_l1 = (float*)carve((size_t)B * 256 * 4);
  float*  n1_l1  = (float*)carve((size_t)B * K * 256 * 4);
  __bf16* Qin    = (__bf16*)carve((size_t)CH * 512 * 2);
  __bf16* Kin    = (__bf16*)carve((size_t)CHK * 768 * 2);
  float*  Qh     = (float*)carve((size_t)CH * 512 * 4);
  float*  Kh     = (float*)carve((size_t)CHK * 512 * 4);
  float*  Vh     = (float*)carve((size_t)CHK * 512 * 4);
  __bf16* Obf    = (__bf16*)carve((size_t)CH * 512 * 2);
  float*  oMul   = (float*)carve((size_t)CH * 4);
  __bf16* Min    = (__bf16*)carve((size_t)CH * 768 * 2);
  __bf16* Hbuf   = (__bf16*)carve((size_t)CH * 256 * 2);

  // ---- stage weights (transpose + bf16) and constants ----
  for (int l = 0; l < 2; ++l) {
    k_convert_wT<<<512, 256, 0, stream>>>(Wq + (size_t)l * 512 * 512, WqT[l], 512, 512);
    k_convert_wT<<<768, 256, 0, stream>>>(Wk + (size_t)l * 768 * 512, WkT[l], 768, 512);
    k_convert_wT<<<768, 256, 0, stream>>>(Wv + (size_t)l * 768 * 512, WvT[l], 768, 512);
    k_convert_wT<<<512, 256, 0, stream>>>(Wo + (size_t)l * 512 * 512, WoT[l], 512, 512);
    k_convert_wT<<<384, 256, 0, stream>>>(W1 + (size_t)l * 768 * 256, W1T[l], 768, 256);
    k_convert_wT<<<128, 256, 0, stream>>>(W2 + (size_t)l * 256 * 256, W2T[l], 256, 256);
  }
  k_te0<<<1, 256, 0, stream>>>(time_b, te0);
  k_gather_feats<<<B, 256, 0, stream>>>(memory, nodef, src_nodes, f_src0, B);
  k_gather_feats<<<B * K, 256, 0, stream>>>(memory, nodef, neighbors1, f_n1_0, B * K);

  // ---- one attn() invocation, chunked over queries ----
  auto run_attn = [&](int layer, int nQtot, const float* srcfeat,
                      int kinMode, const float* directRows,
                      const int* nbrIdx, const int* edgeIdx,
                      const float* edgeTimes, int tsDiv, float* outp) {
    for (int q0 = 0; q0 < nQtot; q0 += CH) {
      k_build_qin<<<CH, 256, 0, stream>>>(srcfeat + (size_t)q0 * 256, te0, Qin);
      k_build_kin<<<CHK, 256, 0, stream>>>(memory, nodef, edgef, time_w, time_b,
          directRows, nbrIdx, edgeIdx, timestamps, edgeTimes,
          tsDiv, kinMode, q0 * K, Kin);
      // Qh = Qin @ Wq + bq            [CH,512] x [512,512]
      k_gemm_bf16<<<dim3(4, CH / 128), 256, 0, stream>>>(
          Qin, 512, WqT[layer], 512, bq + layer * 512, nullptr, Qh, nullptr, 512, 0);
      // Kh/Vh = Kin @ Wk/Wv + b       [CHK,768] x [768,512]
      k_gemm_bf16<<<dim3(4, CHK / 128), 256, 0, stream>>>(
          Kin, 768, WkT[layer], 768, bk + layer * 512, nullptr, Kh, nullptr, 512, 0);
      k_gemm_bf16<<<dim3(4, CHK / 128), 256, 0, stream>>>(
          Kin, 768, WvT[layer], 768, bv + layer * 512, nullptr, Vh, nullptr, 512, 0);
      // masked softmax + weighted V sum
      k_attn_combine<<<(CH * 8 + 255) / 256, 256, 0, stream>>>(
          Qh, Kh, Vh, nbrIdx, q0 * K, Obf, oMul, CH);
      k_fill_min_src<<<CH, 256, 0, stream>>>(srcfeat + (size_t)q0 * 256, Min);
      // Min[:,0:512] = (Obf @ Wo + bo) * allMaskFlag   (bf16 epilogue)
      k_gemm_bf16<<<dim3(4, CH / 128), 256, 0, stream>>>(
          Obf, 512, WoT[layer], 512, bo + layer * 512, oMul,
          nullptr, Min, 768, F_BF16 | F_ROWMUL);
      // H = relu(Min @ W1 + b1)        [CH,768] x [768,256]
      k_gemm_bf16<<<dim3(2, CH / 128), 256, 0, stream>>>(
          Min, 768, W1T[layer], 768, b1 + layer * 256, nullptr,
          nullptr, Hbuf, 256, F_BF16 | F_RELU);
      // out = H @ W2 + b2              [CH,256] x [256,256]
      k_gemm_bf16<<<dim3(2, CH / 128), 256, 0, stream>>>(
          Hbuf, 256, W2T[layer], 256, b2 + layer * 256, nullptr,
          outp + (size_t)q0 * 256, nullptr, 256, 0);
    }
  };

  // layer-1 embedding of sources
  run_attn(0, B, f_src0, 0, nullptr, neighbors1, edge_idx1, edge_times1, 1, src_l1);
  // layer-1 embedding of hop-1 neighbors (hop-2 samples); ts = timestamps[(g/K)/K]
  run_attn(0, B * K, f_n1_0, 0, nullptr, neighbors2, edge_idx2, edge_times2, K, n1_l1);
  // layer-2 embedding of sources (neighbor embeddings come from n1_l1 directly)
  run_attn(1, B, src_l1, 1, n1_l1, neighbors1, edge_idx1, edge_times1, 1, (float*)d_out);
}
